// CA_MultiHead_24799141167540
// MI455X (gfx1250) — matrix operand order repaired and checked by reading, not compiled, not run on hardware
//
#include <hip/hip_runtime.h>

// Cross-attention multi-head for MI455X (gfx1250, wave32, WMMA).
// fp32 inputs -> bf16 Q/K/V via WMMA projections (Q pre-scaled by 256^-0.5)
// -> flash attention (online softmax, 64-key chunks, K/V staged into LDS with
// GLOBAL_LOAD_ASYNC_TO_LDS_B128 / ASYNCcnt) -> bf16 WMMA output projection.
// All matmuls: v_wmma_f32_16x16x32_bf16 (f32 accumulate).

typedef __attribute__((ext_vector_type(16))) __bf16 v16bf;
typedef __attribute__((ext_vector_type(8)))  float  v8f;

struct alignas(16) bf16x8 { __bf16 x[8]; };

static constexpr int S   = 4096;   // Se == Sd
static constexpr int DM  = 256;    // n_emb
static constexpr int HD  = 64;     // head size
static constexpr int NH  = 4;      // heads
static constexpr int NB  = 4;      // batch
static constexpr float SCALE = 0.0625f;   // 256^-0.5 (folded into Q)

#define WMMA_BF16(A_, B_, C_) \
  __builtin_amdgcn_wmma_f32_16x16x32_bf16(false, (A_), false, (B_), (short)0, (C_), false, false)

// A-matrix fragment, 16-bit 16x32 (ISA 7.12.2): lane&15 = M row;
// lanes 0-15 hold K = {0..7, 16..23}, lanes 16-31 K = {8..15, 24..31}.
// Caller passes p already offset by the per-lane kb; runs are p[0..7], p[16..23].
__device__ __forceinline__ v16bf frag_a16(const __bf16* p) {
  bf16x8 lo = *(const bf16x8*)p;
  bf16x8 hi = *(const bf16x8*)(p + 16);
  v16bf a;
#pragma unroll
  for (int i = 0; i < 8; ++i) { a[i] = lo.x[i]; a[i + 8] = hi.x[i]; }
  return a;
}

// B-matrix fragment, 16-bit 32x16: lane&15 = N col; lanes 0-15 hold K = {0..7, 16..23},
// lanes 16-31 K = {8..15, 24..31} (caller applies the +8 offset). Runs p[0..7], p[16..23].
__device__ __forceinline__ v16bf frag_b16(const __bf16* p) {
  bf16x8 lo = *(const bf16x8*)p;
  bf16x8 hi = *(const bf16x8*)(p + 16);
  v16bf b;
#pragma unroll
  for (int i = 0; i < 8; ++i) { b[i] = lo.x[i]; b[i + 8] = hi.x[i]; }
  return b;
}

// A fragment sourced from fp32 memory with on-the-fly bf16 conversion.
__device__ __forceinline__ v16bf frag_a_from_f32(const float* p) {
  float4 f0 = *(const float4*)(p);
  float4 f1 = *(const float4*)(p + 4);
  float4 f2 = *(const float4*)(p + 16);
  float4 f3 = *(const float4*)(p + 20);
  float lo[8] = {f0.x, f0.y, f0.z, f0.w, f1.x, f1.y, f1.z, f1.w};
  float hi[8] = {f2.x, f2.y, f2.z, f2.w, f3.x, f3.y, f3.z, f3.w};
  v16bf a;
#pragma unroll
  for (int i = 0; i < 8; ++i) { a[i] = (__bf16)lo[i]; a[i + 8] = (__bf16)hi[i]; }
  return a;
}

// Async 16B copy global -> LDS (per-lane), tracked by ASYNCcnt.
// VDST carries the workgroup-relative LDS byte offset (low 32 bits of the
// generic shared-aperture pointer, ISA 10.2); VADDR carries the 64-bit address.
__device__ __forceinline__ void async_copy_b128(const void* lds_dst, const void* gsrc) {
  unsigned dst = (unsigned)(size_t)lds_dst;
  asm volatile("global_load_async_to_lds_b128 %0, %1, off"
               :: "v"(dst), "v"(gsrc) : "memory");
}
__device__ __forceinline__ void async_wait0() {
  asm volatile("s_wait_asynccnt 0" ::: "memory");
}

// ---------------------------------------------------------------------------
// Kernel 0: weight conversion. Wq/Wk/Wv [h][k][n] fp32 -> [h][n][k] bf16.
// Wo [o][d] fp32 -> bf16 (already lane=o, k=d contiguous layout).
// ---------------------------------------------------------------------------
__global__ void ca_convert_weights(const float* __restrict__ Wq, const float* __restrict__ Wk,
                                   const float* __restrict__ Wv, const float* __restrict__ Wo,
                                   __bf16* __restrict__ Wqt, __bf16* __restrict__ Wkt,
                                   __bf16* __restrict__ Wvt, __bf16* __restrict__ Wot) {
  int i = blockIdx.x * 256 + threadIdx.x;          // 0 .. 65535
  int h = i >> 14, rem = i & 16383;
  int k = rem >> 6, n = rem & 63;
  int dst = h * (HD * DM) + n * DM + k;
  Wqt[dst] = (__bf16)Wq[i];
  Wkt[dst] = (__bf16)Wk[i];
  Wvt[dst] = (__bf16)Wv[i];
  Wot[i]   = (__bf16)Wo[i];
}

// ---------------------------------------------------------------------------
// Kernel 1: Q/K/V projections. grid = (S/128, B*H, 3); 8 waves/block, one
// 16x64 output tile per wave, K=256 contraction (8 WMMA k-chunks x 4 n-tiles).
// Q output is pre-multiplied by SCALE. z==2 (V) stores transposed [bh][e][s].
// ---------------------------------------------------------------------------
__global__ void ca_proj(const float* __restrict__ x_dec, const float* __restrict__ x_enc,
                        const __bf16* __restrict__ Wqt, const __bf16* __restrict__ Wkt,
                        const __bf16* __restrict__ Wvt,
                        __bf16* __restrict__ Qw, __bf16* __restrict__ Kw,
                        __bf16* __restrict__ Vtw) {
  const int lane = threadIdx.x & 31, wave = threadIdx.x >> 5;
  const int z  = blockIdx.z;
  const int bh = blockIdx.y, b = bh >> 2, h = bh & 3;
  const int m0 = (blockIdx.x * 8 + wave) * 16;
  const int ml = lane & 15;
  const int hiA = (lane & 16) ? 8 : 0;
  const int hiB = (lane & 16) ? 8 : 0;

  const float*  x  = (z == 0) ? x_dec : x_enc;
  const __bf16* Wt = (z == 0) ? Wqt : (z == 1) ? Wkt : Wvt;
  const float*  xrow  = x + ((size_t)b * S + m0 + ml) * DM;
  const __bf16* wbase = Wt + (size_t)h * HD * DM;     // [n][k]

  v8f z8 = {0.f, 0.f, 0.f, 0.f, 0.f, 0.f, 0.f, 0.f};
  v8f acc[4] = {z8, z8, z8, z8};

#pragma unroll
  for (int kc = 0; kc < DM; kc += 32) {
    v16bf a = frag_a_from_f32(xrow + kc + hiA);
#pragma unroll
    for (int nt = 0; nt < 4; ++nt) {
      v16bf bf = frag_b16(wbase + (size_t)(nt * 16 + ml) * DM + kc + hiB);
      acc[nt] = WMMA_BF16(a, bf, acc[nt]);
    }
  }

  const float qs = (z == 0) ? SCALE : 1.0f;     // fold attention scale into Q
  if (z < 2) {
    __bf16* o = ((z == 0) ? Qw : Kw) + (size_t)bh * S * HD;
#pragma unroll
    for (int nt = 0; nt < 4; ++nt)
#pragma unroll
      for (int r = 0; r < 8; ++r) {
        int row = m0 + r + ((lane & 16) ? 8 : 0);
        o[(size_t)row * HD + nt * 16 + ml] = (__bf16)(acc[nt][r] * qs);
      }
  } else {
    __bf16* o = Vtw + (size_t)bh * HD * S;              // [e][s]
#pragma unroll
    for (int nt = 0; nt < 4; ++nt)
#pragma unroll
      for (int r = 0; r < 8; ++r) {
        int row = m0 + r + ((lane & 16) ? 8 : 0);
        int e = nt * 16 + ml;
        o[(size_t)e * S + row] = (__bf16)acc[nt][r];
      }
  }
}

// ---------------------------------------------------------------------------
// Kernel 2: flash attention. grid = (Sd/128, B*H); 8 waves/block, each wave
// owns 16 queries; 64-key chunks staged via async global->LDS copies; online
// softmax with shfl-based row reductions; P reshaped C->A layout via LDS.
// Per iteration: 16 WMMAs (8 score + 8 P@V), fixed softmax cost amortized
// over 64 keys.
// ---------------------------------------------------------------------------
__global__ void ca_flash(const __bf16* __restrict__ Qw, const __bf16* __restrict__ Kw,
                         const __bf16* __restrict__ Vtw, __bf16* __restrict__ Ocat) {
  __shared__ __align__(16) __bf16 Kc[64 * 64];       // [key][d]
  __shared__ __align__(16) __bf16 Vc[64 * 64];       // [e][key]
  __shared__ __align__(16) __bf16 Pb[8][16 * 64];    // per-wave P tile [m][n]

  const int tid = threadIdx.x, lane = tid & 31, wave = tid >> 5;
  const int bh = blockIdx.y, b = bh >> 2, h = bh & 3;
  const int m0 = blockIdx.x * 128 + wave * 16;
  const int ml = lane & 15;
  const int hiA = (lane & 16) ? 8 : 0;
  const int hiB = (lane & 16) ? 8 : 0;
  const int hi8 = (lane & 16) ? 8 : 0;

  // Q fragments (reused across the whole key loop); Q already carries SCALE.
  const __bf16* qrow = Qw + ((size_t)bh * S + m0 + ml) * HD;
  v16bf aq0 = frag_a16(qrow + 0  + hiA);
  v16bf aq1 = frag_a16(qrow + 32 + hiA);

  const __bf16* kbase = Kw  + (size_t)bh * S * HD;   // [key][d]
  const __bf16* vbase = Vtw + (size_t)bh * HD * S;   // [e][key]

  v8f z8 = {0.f, 0.f, 0.f, 0.f, 0.f, 0.f, 0.f, 0.f};
  v8f acc[4] = {z8, z8, z8, z8};
  float mr[8], lr[8];
#pragma unroll
  for (int r = 0; r < 8; ++r) { mr[r] = -3.0e38f; lr[r] = 0.f; }

  for (int j = 0; j < S; j += 64) {
    __syncthreads();
    // Stage K chunk (64 keys x 64 d) and V^T chunk (64 e x 64 keys), 8KB each,
    // as per-lane async 16B transfers: 4 issues/thread, ASYNCcnt-tracked.
#pragma unroll
    for (int q2 = 0; q2 < 2; ++q2) {
      int i = tid + q2 * 256;                 // 0..511
      int row = i >> 3, part = i & 7;         // row: key or e; part*8 elems
      async_copy_b128(&Kc[row * 64 + part * 8],
                      kbase + (size_t)(j + row) * HD + part * 8);
      async_copy_b128(&Vc[row * 64 + part * 8],
                      vbase + (size_t)row * S + j + part * 8);
    }
    async_wait0();
    __syncthreads();

    // scores: 16 queries x 64 keys, contraction over d=64 (2 WMMA per subtile)
    v8f sc[4] = {z8, z8, z8, z8};
#pragma unroll
    for (int st = 0; st < 4; ++st) {
      v16bf b0 = frag_b16(&Kc[(size_t)(st * 16 + ml) * 64 + 0  + hiB]);
      v16bf b1 = frag_b16(&Kc[(size_t)(st * 16 + ml) * 64 + 32 + hiB]);
      sc[st] = WMMA_BF16(aq0, b0, sc[st]);
      sc[st] = WMMA_BF16(aq1, b1, sc[st]);
    }

    // online softmax; rows live across 16-lane halves in C layout
#pragma unroll
    for (int r = 0; r < 8; ++r) {
      float x0 = sc[0][r], x1 = sc[1][r], x2 = sc[2][r], x3 = sc[3][r];
      float mx = fmaxf(fmaxf(x0, x1), fmaxf(x2, x3));
#pragma unroll
      for (int msk = 1; msk < 16; msk <<= 1) mx = fmaxf(mx, __shfl_xor(mx, msk, 16));
      float mn = fmaxf(mr[r], mx);
      float p0 = __expf(x0 - mn), p1 = __expf(x1 - mn);
      float p2 = __expf(x2 - mn), p3 = __expf(x3 - mn);
      float sum = (p0 + p1) + (p2 + p3);
#pragma unroll
      for (int msk = 1; msk < 16; msk <<= 1) sum += __shfl_xor(sum, msk, 16);
      float corr = __expf(mr[r] - mn);
      lr[r] = lr[r] * corr + sum;
      mr[r] = mn;
      acc[0][r] *= corr; acc[1][r] *= corr; acc[2][r] *= corr; acc[3][r] *= corr;
      int base = (r + hi8) * 64 + ml;
      Pb[wave][base]      = (__bf16)p0;
      Pb[wave][base + 16] = (__bf16)p1;
      Pb[wave][base + 32] = (__bf16)p2;
      Pb[wave][base + 48] = (__bf16)p3;
    }

    // O += P (16x64) @ V (64x64): 2 k-chunks x 4 e-tiles
    v16bf ap0 = frag_a16(&Pb[wave][ml * 64 + 0  + hiA]);
    v16bf ap1 = frag_a16(&Pb[wave][ml * 64 + 32 + hiA]);
#pragma unroll
    for (int t = 0; t < 4; ++t) {
      v16bf bv0 = frag_b16(&Vc[(size_t)(t * 16 + ml) * 64 + 0  + hiB]);
      v16bf bv1 = frag_b16(&Vc[(size_t)(t * 16 + ml) * 64 + 32 + hiB]);
      acc[t] = WMMA_BF16(ap0, bv0, acc[t]);
      acc[t] = WMMA_BF16(ap1, bv1, acc[t]);
    }
  }

  // normalize and write concat-head output [b][s][h*64+e] in bf16
#pragma unroll
  for (int r = 0; r < 8; ++r) {
    float inv = 1.0f / lr[r];
    int row = m0 + r + hi8;
    __bf16* op = Ocat + ((size_t)b * S + row) * DM + h * HD;
#pragma unroll
    for (int t = 0; t < 4; ++t)
      op[t * 16 + ml] = (__bf16)((float)acc[t][r] * inv);
  }
}

// ---------------------------------------------------------------------------
// Kernel 3: output projection. out[s][o] = sum_d Ocat[s][d] * Wo[o][d].
// grid = (B*Sd/128, 4); each wave: 16 rows x 64 cols, K=256.
// ---------------------------------------------------------------------------
__global__ void ca_outproj(const __bf16* __restrict__ Ocat, const __bf16* __restrict__ Wot,
                           float* __restrict__ out) {
  const int lane = threadIdx.x & 31, wave = threadIdx.x >> 5;
  const int m0 = blockIdx.x * 128 + wave * 16;      // flat row in [0, B*S)
  const int oc0 = blockIdx.y * 64;
  const int ml = lane & 15;
  const int hiA = (lane & 16) ? 8 : 0;
  const int hiB = (lane & 16) ? 8 : 0;

  const __bf16* arow = Ocat + ((size_t)m0 + ml) * DM;
  v8f z8 = {0.f, 0.f, 0.f, 0.f, 0.f, 0.f, 0.f, 0.f};
  v8f acc[4] = {z8, z8, z8, z8};

#pragma unroll
  for (int kc = 0; kc < DM; kc += 32) {
    v16bf a = frag_a16(arow + kc + hiA);
#pragma unroll
    for (int t = 0; t < 4; ++t) {
      v16bf bf = frag_b16(Wot + (size_t)(oc0 + t * 16 + ml) * DM + kc + hiB);
      acc[t] = WMMA_BF16(a, bf, acc[t]);
    }
  }
#pragma unroll
  for (int r = 0; r < 8; ++r) {
    int row = m0 + r + ((lane & 16) ? 8 : 0);
#pragma unroll
    for (int t = 0; t < 4; ++t)
      out[(size_t)row * DM + oc0 + t * 16 + ml] = acc[t][r];
  }
}

// ---------------------------------------------------------------------------
extern "C" void kernel_launch(void* const* d_in, const int* in_sizes, int n_in,
                              void* d_out, int out_size, void* d_ws, size_t ws_size,
                              hipStream_t stream) {
  const float* x_enc = (const float*)d_in[0];
  const float* x_dec = (const float*)d_in[1];
  const float* Wq    = (const float*)d_in[2];
  const float* Wk    = (const float*)d_in[3];
  const float* Wv    = (const float*)d_in[4];
  const float* Wo    = (const float*)d_in[5];
  float* out = (float*)d_out;

  // carve workspace (all offsets 256B-aligned; total ~32.5 MB)
  char* p = (char*)d_ws;
  auto carve = [&](size_t bytes) {
    char* r = p;
    p += (bytes + 255) & ~(size_t)255;
    return r;
  };
  __bf16* Wqt  = (__bf16*)carve((size_t)NH * HD * DM * 2);
  __bf16* Wkt  = (__bf16*)carve((size_t)NH * HD * DM * 2);
  __bf16* Wvt  = (__bf16*)carve((size_t)NH * HD * DM * 2);
  __bf16* Wot  = (__bf16*)carve((size_t)DM * DM * 2);
  __bf16* Qw   = (__bf16*)carve((size_t)NB * NH * S * HD * 2);
  __bf16* Kw   = (__bf16*)carve((size_t)NB * NH * S * HD * 2);
  __bf16* Vtw  = (__bf16*)carve((size_t)NB * NH * HD * S * 2);
  __bf16* Ocat = (__bf16*)carve((size_t)NB * S * DM * 2);
  (void)ws_size; (void)in_sizes; (void)n_in; (void)out_size;

  ca_convert_weights<<<dim3(256), 256, 0, stream>>>(Wq, Wk, Wv, Wo, Wqt, Wkt, Wvt, Wot);
  ca_proj<<<dim3(S / 128, NB * NH, 3), 256, 0, stream>>>(x_dec, x_enc, Wqt, Wkt, Wvt,
                                                         Qw, Kw, Vtw);
  ca_flash<<<dim3(S / 128, NB * NH), 256, 0, stream>>>(Qw, Kw, Vtw, Ocat);
  ca_outproj<<<dim3(NB * S / 128, 4), 256, 0, stream>>>(Ocat, Wot, out);
}
